// OPTmodl_77017353552182
// MI455X (gfx1250) — compile-verified, matrix operand order checked by reading, not run on hardware
//
#include <hip/hip_runtime.h>

// ---------------- problem constants ----------------
#define NN      160            // image size
#define AAN     160            // n_angles
#define DDN     226            // det_count
#define CH      64             // conv channels
#define LAM     0.05f
#define PIXELS  (NN*NN)
#define BATCH   2
#define PI_F    3.14159265358979323846f

typedef _Float16 f16;
typedef __attribute__((ext_vector_type(16))) _Float16 v16h;
typedef __attribute__((ext_vector_type(8)))  _Float16 v8h;
typedef __attribute__((ext_vector_type(8)))  float    v8f;
typedef __attribute__((ext_vector_type(4)))  unsigned int v4u;
typedef __attribute__((ext_vector_type(8)))  int v8i;
typedef __attribute__((ext_vector_type(4)))  int v4i;

// ================= trig tables =================
__global__ void k_init_tables(float* cosv, float* sinv) {
    int i = blockIdx.x * blockDim.x + threadIdx.x;
    if (i < AAN) {
        float ang = (2.0f * PI_F / AAN) * (float)i;
        cosv[i] = cosf(ang);
        sinv[i] = sinf(ang);
    }
}

// ================= weight packing for WMMA B-fragments =================
// fragment = 32x16 (K x N) f16; lane l: n = l&15, hi = l>>4 ;
// half h: K = hi*8 + h + (h>=8 ? 8 : 0). Fragment stride 512 halfs,
// lane's 16 halfs contiguous at lane*16.
__global__ void k_pack_w(const float* __restrict__ w_mid, f16* __restrict__ packed) {
    int i = blockIdx.x * blockDim.x + threadIdx.x;
    if (i >= 3 * 9 * 2 * 4 * 32) return;
    int lane = i & 31;  i >>= 5;
    int ct   = i & 3;   i >>= 2;
    int k2   = i & 1;   i >>= 1;
    int tap  = i % 9;
    int layer = i / 9;
    int n  = lane & 15;
    int hi = lane >> 4;
    int frag = ((layer * 9 + tap) * 2 + k2) * 4 + ct;
    f16* dst = packed + frag * 512 + lane * 16;
    int co = ct * 16 + n;
    #pragma unroll
    for (int h = 0; h < 16; ++h) {
        int K  = hi * 8 + h + ((h >= 8) ? 8 : 0);
        int ci = k2 * 32 + K;
        dst[h] = (f16)w_mid[((layer * 64 + co) * 64 + ci) * 9 + tap];
    }
}

// ================= conv1: 1 -> 64, relu, NHWC f16 out =================
__global__ void k_conv1(const float* __restrict__ x, const float* __restrict__ w1,
                        const float* __restrict__ b1, f16* __restrict__ out) {
    int idx = blockIdx.x * blockDim.x + threadIdx.x;
    if (idx >= BATCH * PIXELS * CH) return;
    int co  = idx & 63;
    int pix = (idx >> 6) % PIXELS;
    int img = idx / (PIXELS * CH);
    int y = pix / NN, xx = pix % NN;
    const float* im = x + img * PIXELS;
    float s = 0.f;
    #pragma unroll
    for (int ky = 0; ky < 3; ++ky) {
        int iy = y + ky - 1;
        if (iy < 0 || iy >= NN) continue;
        #pragma unroll
        for (int kx = 0; kx < 3; ++kx) {
            int ix = xx + kx - 1;
            if (ix < 0 || ix >= NN) continue;
            s += im[iy * NN + ix] * w1[co * 9 + ky * 3 + kx];
        }
    }
    s += b1[co];
    s = s > 0.f ? s : 0.f;
    out[(img * PIXELS + pix) * CH + co] = (f16)s;
}

// ================= mid conv: 64 -> 64 via WMMA, relu =================
// grid = (NN rows, BATCH), block = 320 threads = 10 waves, wave w -> pixels [16w,16w+16)
// Input tile (3 rows x 162 cols x 64ch f16) staged in LDS; interior rows fetched
// by the Tensor Data Mover (one D# per image row), halo columns always zero.
__global__ void __launch_bounds__(320)
k_conv_wmma(const f16* __restrict__ in, f16* __restrict__ out,
            const f16* __restrict__ wp, const float* __restrict__ bias) {
    __shared__ __align__(16) f16 tile[3 * 162 * CH];   // 62208 B
    int y   = blockIdx.x;
    int img = blockIdx.y;
    int tid = threadIdx.x;
    unsigned int* t32 = (unsigned int*)tile;

#if __has_builtin(__builtin_amdgcn_tensor_load_to_lds) && __has_builtin(__builtin_amdgcn_s_wait_tensorcnt)
    // ---- TDM path: 3 row-descriptors, cols 0..159 -> tile[r][1..160][*] ----
    if (tid < 32) {
        unsigned lbase = (unsigned)(unsigned long long)(__SIZE_TYPE__)(void*)tile;
        #pragma unroll
        for (int r = 0; r < 3; ++r) {
            int gy = y - 1 + r;   // may be -1 / NN: stays inside workspace, zeroed below
            const f16* rowp = in + (long long)(img * NN + gy) * (NN * CH);
            unsigned long long ga = (unsigned long long)(__SIZE_TYPE__)(const void*)rowp;
            unsigned laddr = lbase + (unsigned)((r * 162 + 1) * CH * 2);
            // ---- D# group 0: count=1 | lds_addr | global_addr | type=2 ----
            v4u g0;
            g0.x = 0x1u;
            g0.y = laddr;
            g0.z = (unsigned)(ga & 0xFFFFFFFFull);
            g0.w = (unsigned)((ga >> 32) & 0x1FFFFFFull) | 0x80000000u;
            // ---- D# group 1: data_size=2B, tensor_dim0=tile_dim0=NN*CH,
            //      tensor_dim1=NN, tile_dim1=1, dim0_stride=NN*CH ----
            const unsigned TD0 = NN * CH;          // 10240 halfs per row
            v8i g1;
            g1[0] = (int)(1u << 16);                               // data_size=1 (2 bytes)
            g1[1] = (int)((TD0 & 0xFFFFu) << 16);                  // tensor_dim0[15:0]
            g1[2] = (int)((TD0 >> 16) | ((unsigned)NN << 16));     // tensor_dim0[31:16] | tensor_dim1[15:0]
            g1[3] = (int)(((unsigned)NN >> 16) | (TD0 << 16));     // tensor_dim1[31:16] | tile_dim0
            g1[4] = 1;                                             // tile_dim1=1, tile_dim2=0
            g1[5] = (int)TD0;                                      // tensor_dim0_stride[31:0]
            g1[6] = 0;                                             // stride0[47:32] | stride1[15:0]
            g1[7] = 0;                                             // stride1[47:16]
            v4i z4 = {};
#if defined(__clang_major__) && __clang_major__ >= 23
            v8i z8 = {};
            __builtin_amdgcn_tensor_load_to_lds(g0, g1, z4, z4, z8, 0);
#else
            __builtin_amdgcn_tensor_load_to_lds(g0, g1, z4, z4, 0);
#endif
        }
    }
    // halo columns gx=-1 and gx=160 are always outside the image -> zero
    if (tid < 192) {
        int d = tid & 31;                 // dword within channel vector
        int q = tid >> 5;                 // 0..5
        int r = q >> 1;
        int c = (q & 1) ? 161 : 0;
        t32[(r * 162 + c) * 32 + d] = 0u;
    }
    if (tid < 32) __builtin_amdgcn_s_wait_tensorcnt(0);
    __syncthreads();
    if (y == 0 || y == NN - 1) {          // zero the out-of-image row
        int r = (y == 0) ? 0 : 2;
        for (int i = tid; i < NN * (CH / 2); i += 320)
            t32[(r * 162 + 1) * 32 + i] = 0u;
        __syncthreads();
    }
#else
    // ---- fallback: cooperative dword load with zero-padded halo ----
    const int ndw = 3 * 162 * (CH / 2);
    for (int i = tid; i < ndw; i += 320) {
        int d = i & 31;
        int c = (i >> 5) % 162;
        int r = (i >> 5) / 162;
        int gy = y - 1 + r, gx = c - 1;
        unsigned int v = 0;
        if (gy >= 0 && gy < NN && gx >= 0 && gx < NN)
            v = *(const unsigned int*)(in + ((img * NN + gy) * NN + gx) * CH + d * 2);
        t32[i] = v;
    }
    __syncthreads();
#endif

    int lane = tid & 31;
    int wv   = tid >> 5;
    int m    = lane & 15;
    int hi   = lane >> 4;
    int x0   = wv * 16;

    v8f acc[4] = {};

    #pragma unroll
    for (int ky = 0; ky < 3; ++ky) {
        #pragma unroll
        for (int kx = 0; kx < 3; ++kx) {
            int tap = ky * 3 + kx;
            const f16* arow = tile + (ky * 162 + (x0 + m + kx)) * CH;
            #pragma unroll
            for (int k2 = 0; k2 < 2; ++k2) {
                // A fragment: K = k2*32 + hi*8 + h (h<8) / +16 (h>=8) -> two b128 LDS loads
                const f16* ab = arow + k2 * 32 + hi * 8;
                v8h alo = *(const v8h*)(ab);
                v8h ahi = *(const v8h*)(ab + 16);
                v16h a;
                #pragma unroll
                for (int h = 0; h < 8; ++h) { a[h] = alo[h]; a[h + 8] = ahi[h]; }
                const f16* wb = wp + ((__SIZE_TYPE__)(tap * 2 + k2) * 4) * 512 + lane * 16;
                #pragma unroll
                for (int ct = 0; ct < 4; ++ct) {
                    v16h b = *(const v16h*)(wb + ct * 512);
                    acc[ct] = __builtin_amdgcn_wmma_f32_16x16x32_f16(
                        false, a, false, b, (short)0, acc[ct], false, false);
                }
            }
        }
    }

    // store: acc[ct] element e -> pixel x0 + hi*8 + e, channel ct*16 + (lane&15)
    float bv[4];
    #pragma unroll
    for (int ct = 0; ct < 4; ++ct) bv[ct] = bias[ct * 16 + (lane & 15)];
    #pragma unroll
    for (int ct = 0; ct < 4; ++ct) {
        #pragma unroll
        for (int e = 0; e < 8; ++e) {
            int px = x0 + hi * 8 + e;
            float v = acc[ct][e] + bv[ct];
            v = v > 0.f ? v : 0.f;
            out[((img * NN + y) * NN + px) * CH + ct * 16 + (lane & 15)] = (f16)v;
        }
    }
}

// ====== last conv 64->1 (no relu) + residual + rhs = atb/LAM + z ======
__global__ void k_convlast_rhs(const f16* __restrict__ h, const float* __restrict__ w_last,
                               const float* __restrict__ b_last, const float* __restrict__ xin,
                               const float* __restrict__ atb, float* __restrict__ rhs) {
    int idx = blockIdx.x * blockDim.x + threadIdx.x;
    if (idx >= BATCH * PIXELS) return;
    int img = idx / PIXELS;
    int pix = idx % PIXELS;
    int y = pix / NN, xx = pix % NN;
    float s = 0.f;
    for (int ky = 0; ky < 3; ++ky) {
        int iy = y + ky - 1;
        if (iy < 0 || iy >= NN) continue;
        for (int kx = 0; kx < 3; ++kx) {
            int ix = xx + kx - 1;
            if (ix < 0 || ix >= NN) continue;
            const f16* v = h + ((img * NN + iy) * NN + ix) * CH;
            int tap = ky * 3 + kx;
            #pragma unroll
            for (int ci = 0; ci < CH; ++ci)
                s += (float)v[ci] * w_last[ci * 9 + tap];
        }
    }
    rhs[idx] = s + b_last[0] + xin[idx] + atb[idx] * (1.0f / LAM);
}

// ================= radon forward: sino = sum(bilinear)/N =================
__device__ __forceinline__ float bilin(const float* __restrict__ im, float x, float y) {
    float x0 = floorf(x), y0 = floorf(y);
    int ix = (int)x0, iy = (int)y0;
    float wx = x - x0, wy = y - y0;
    float v00 = (ix >= 0 && ix < NN && iy >= 0 && iy < NN)         ? im[iy * NN + ix]           : 0.f;
    float v01 = (ix + 1 >= 0 && ix + 1 < NN && iy >= 0 && iy < NN) ? im[iy * NN + ix + 1]       : 0.f;
    float v10 = (ix >= 0 && ix < NN && iy + 1 >= 0 && iy + 1 < NN) ? im[(iy + 1) * NN + ix]     : 0.f;
    float v11 = (ix + 1 >= 0 && ix + 1 < NN && iy + 1 >= 0 && iy + 1 < NN) ? im[(iy + 1) * NN + ix + 1] : 0.f;
    return (1.f - wx) * (1.f - wy) * v00 + wx * (1.f - wy) * v01
         + (1.f - wx) * wy * v10 + wx * wy * v11;
}

__global__ void k_radon(const float* __restrict__ p, const float* __restrict__ cosv,
                        const float* __restrict__ sinv, float* __restrict__ sino) {
    int idx = blockIdx.x * blockDim.x + threadIdx.x;
    if (idx >= BATCH * AAN * DDN) return;
    int img = idx / (AAN * DDN);
    int a   = (idx / DDN) % AAN;
    int s   = idx % DDN;
    float cs = cosv[a], sn = sinv[a];
    float sc = (float)s - (DDN - 1) * 0.5f;
    const float* im = p + img * PIXELS;
    float sum = 0.f;
    for (int t = 0; t < DDN; ++t) {
        float tc = (float)t - (DDN - 1) * 0.5f;
        float px = 79.5f + sc * cs - tc * sn;
        float py = 79.5f + sc * sn + tc * cs;
        sum += bilin(im, px, py);
    }
    sino[(img * AAN + a) * DDN + s] = sum * (1.0f / NN);
}

// ======= backproject + finish AtA: Ap = bp(sino)*(pi/A)/LAM + p =======
__global__ void k_backproj(const float* __restrict__ sino, const float* __restrict__ p,
                           const float* __restrict__ cosv, const float* __restrict__ sinv,
                           float* __restrict__ Ap) {
    int idx = blockIdx.x * blockDim.x + threadIdx.x;
    if (idx >= BATCH * PIXELS) return;
    int img = idx / PIXELS;
    int pix = idx % PIXELS;
    float xc = (float)(pix % NN) - 79.5f;
    float yc = (float)(pix / NN) - 79.5f;
    const float* sg = sino + img * AAN * DDN;
    float sum = 0.f;
    for (int a = 0; a < AAN; ++a) {
        float U = xc * cosv[a] + yc * sinv[a] + (DDN - 1) * 0.5f;
        float f = floorf(U);
        int i0 = (int)f;
        float w = U - f;
        float v0 = (i0 >= 0 && i0 < DDN)         ? sg[a * DDN + i0]     : 0.f;
        float v1 = (i0 + 1 >= 0 && i0 + 1 < DDN) ? sg[a * DDN + i0 + 1] : 0.f;
        sum += (1.f - w) * v0 + w * v1;
    }
    Ap[idx] = sum * (PI_F / AAN) * (1.0f / LAM) + p[idx];
}

// ================= CG helpers (deterministic tree reductions) =================
__global__ void k_cg_init(const float* __restrict__ rhs, float* __restrict__ x,
                          float* __restrict__ r, float* __restrict__ p) {
    int i = blockIdx.x * blockDim.x + threadIdx.x;
    if (i >= BATCH * PIXELS) return;
    x[i] = 0.f;
    float v = rhs[i];
    r[i] = v;
    p[i] = v;
}

__global__ void k_dot(const float* __restrict__ u, const float* __restrict__ v,
                      float* __restrict__ partials) {
    __shared__ float red[256];
    int img = blockIdx.y;
    const float* a = u + img * PIXELS;
    const float* b = v + img * PIXELS;
    float s = 0.f;
    for (int i = blockIdx.x * 256 + threadIdx.x; i < PIXELS; i += 64 * 256)
        s += a[i] * b[i];
    red[threadIdx.x] = s;
    __syncthreads();
    for (int t = 128; t > 0; t >>= 1) {
        if (threadIdx.x < t) red[threadIdx.x] += red[threadIdx.x + t];
        __syncthreads();
    }
    if (threadIdx.x == 0) partials[img * 64 + blockIdx.x] = red[0];
}

__global__ void k_scalar_init(const float* __restrict__ partials, float* __restrict__ rTr) {
    int img = threadIdx.x;
    if (img >= BATCH) return;
    float s = 0.f;
    for (int i = 0; i < 64; ++i) s += partials[img * 64 + i];
    rTr[img] = s;
}

__global__ void k_scalarA(const float* __restrict__ partials, const float* __restrict__ rTr,
                          float* __restrict__ alpha, int* __restrict__ active) {
    int img = threadIdx.x;
    if (img >= BATCH) return;
    float pAp = 0.f;
    for (int i = 0; i < 64; ++i) pAp += partials[img * 64 + i];
    active[img] = (rTr[img] >= 1e-6f) ? 1 : 0;
    alpha[img]  = rTr[img] / pAp;
}

__global__ void k_update_xr(float* __restrict__ x, float* __restrict__ r,
                            const float* __restrict__ p, const float* __restrict__ Ap,
                            const float* __restrict__ alpha, const int* __restrict__ active,
                            float* __restrict__ partials) {
    __shared__ float red[256];
    int img = blockIdx.y;
    float al = alpha[img];
    int act = active[img];
    int base = img * PIXELS;
    float s = 0.f;
    for (int i = blockIdx.x * 256 + threadIdx.x; i < PIXELS; i += 64 * 256) {
        float rn = r[base + i];
        if (act) {
            x[base + i] += al * p[base + i];
            rn -= al * Ap[base + i];
            r[base + i] = rn;
        }
        s += rn * rn;
    }
    red[threadIdx.x] = s;
    __syncthreads();
    for (int t = 128; t > 0; t >>= 1) {
        if (threadIdx.x < t) red[threadIdx.x] += red[threadIdx.x + t];
        __syncthreads();
    }
    if (threadIdx.x == 0) partials[img * 64 + blockIdx.x] = red[0];
}

__global__ void k_scalarB(const float* __restrict__ partials, float* __restrict__ rTr,
                          const int* __restrict__ active, float* __restrict__ beta) {
    int img = threadIdx.x;
    if (img >= BATCH) return;
    float rn = 0.f;
    for (int i = 0; i < 64; ++i) rn += partials[img * 64 + i];
    if (active[img]) {
        beta[img] = rn / rTr[img];
        rTr[img]  = rn;
    } else {
        beta[img] = 0.f;
    }
}

__global__ void k_update_p(float* __restrict__ p, const float* __restrict__ r,
                           const float* __restrict__ beta, const int* __restrict__ active) {
    int i = blockIdx.x * blockDim.x + threadIdx.x;
    if (i >= BATCH * PIXELS) return;
    int img = i / PIXELS;
    if (!active[img]) return;
    p[i] = r[i] + beta[img] * p[i];
}

__global__ void k_copy(float* __restrict__ dst, const float* __restrict__ src, int n) {
    int i = blockIdx.x * blockDim.x + threadIdx.x;
    if (i < n) dst[i] = src[i];
}

// ================= host orchestration =================
extern "C" void kernel_launch(void* const* d_in, const int* in_sizes, int n_in,
                              void* d_out, int out_size, void* d_ws, size_t ws_size,
                              hipStream_t stream) {
    (void)in_sizes; (void)n_in; (void)out_size; (void)ws_size;
    const float* atb    = (const float*)d_in[0];
    const float* w1     = (const float*)d_in[1];
    const float* b1     = (const float*)d_in[2];
    const float* w_mid  = (const float*)d_in[3];
    const float* b_mid  = (const float*)d_in[4];
    const float* w_last = (const float*)d_in[5];
    const float* b_last = (const float*)d_in[6];
    float* out = (float*)d_out;

    char* ws = (char*)d_ws;
    size_t off = 0;
    auto take = [&](size_t bytes) -> char* {
        char* p = ws + off;
        off += (bytes + 255) & ~(size_t)255;
        return p;
    };
    float* cosv    = (float*)take(AAN * sizeof(float));
    float* sinv    = (float*)take(AAN * sizeof(float));
    f16*   wpack   = (f16*)  take((size_t)3 * 9 * 2 * 4 * 512 * sizeof(f16));
    f16*   h0      = (f16*)  take((size_t)BATCH * PIXELS * CH * sizeof(f16));
    f16*   h1      = (f16*)  take((size_t)BATCH * PIXELS * CH * sizeof(f16));
    float* xbuf    = (float*)take((size_t)BATCH * PIXELS * sizeof(float));
    float* rhs     = (float*)take((size_t)BATCH * PIXELS * sizeof(float));
    float* rbuf    = (float*)take((size_t)BATCH * PIXELS * sizeof(float));
    float* pbuf    = (float*)take((size_t)BATCH * PIXELS * sizeof(float));
    float* Apbuf   = (float*)take((size_t)BATCH * PIXELS * sizeof(float));
    float* sino    = (float*)take((size_t)BATCH * AAN * DDN * sizeof(float));
    float* partials= (float*)take((size_t)BATCH * 64 * sizeof(float));
    float* rTr     = (float*)take(BATCH * sizeof(float));
    float* alpha   = (float*)take(BATCH * sizeof(float));
    float* beta    = (float*)take(BATCH * sizeof(float));
    int*   active  = (int*)  take(BATCH * sizeof(int));

    const int TPB = 256;
    const int nPix = BATCH * PIXELS;

    k_init_tables<<<1, 256, 0, stream>>>(cosv, sinv);
    k_pack_w<<<(3 * 9 * 2 * 4 * 32 + TPB - 1) / TPB, TPB, 0, stream>>>(w_mid, wpack);
    k_copy<<<(nPix + TPB - 1) / TPB, TPB, 0, stream>>>(xbuf, atb, nPix);

    for (int k = 0; k < 3; ++k) {
        // ---- dw_forward ----
        k_conv1<<<(nPix * CH + TPB - 1) / TPB, TPB, 0, stream>>>(xbuf, w1, b1, h0);
        // layer 0: h0 -> h1 ; layer 1: h1 -> h0 ; layer 2: h0 -> h1
        k_conv_wmma<<<dim3(NN, BATCH), 320, 0, stream>>>(h0, h1, wpack + (size_t)0 * 9 * 2 * 4 * 512, b_mid + 0 * 64);
        k_conv_wmma<<<dim3(NN, BATCH), 320, 0, stream>>>(h1, h0, wpack + (size_t)1 * 9 * 2 * 4 * 512, b_mid + 1 * 64);
        k_conv_wmma<<<dim3(NN, BATCH), 320, 0, stream>>>(h0, h1, wpack + (size_t)2 * 9 * 2 * 4 * 512, b_mid + 2 * 64);
        k_convlast_rhs<<<(nPix + TPB - 1) / TPB, TPB, 0, stream>>>(h1, w_last, b_last, xbuf, atb, rhs);

        // ---- dc: 5-step CG on AtA, x starts at 0 (reuses xbuf) ----
        k_cg_init<<<(nPix + TPB - 1) / TPB, TPB, 0, stream>>>(rhs, xbuf, rbuf, pbuf);
        k_dot<<<dim3(64, BATCH), 256, 0, stream>>>(rhs, rhs, partials);
        k_scalar_init<<<1, BATCH, 0, stream>>>(partials, rTr);
        for (int step = 0; step < 5; ++step) {
            k_radon<<<(BATCH * AAN * DDN + TPB - 1) / TPB, TPB, 0, stream>>>(pbuf, cosv, sinv, sino);
            k_backproj<<<(nPix + TPB - 1) / TPB, TPB, 0, stream>>>(sino, pbuf, cosv, sinv, Apbuf);
            k_dot<<<dim3(64, BATCH), 256, 0, stream>>>(pbuf, Apbuf, partials);
            k_scalarA<<<1, BATCH, 0, stream>>>(partials, rTr, alpha, active);
            k_update_xr<<<dim3(64, BATCH), 256, 0, stream>>>(xbuf, rbuf, pbuf, Apbuf, alpha, active, partials);
            k_scalarB<<<1, BATCH, 0, stream>>>(partials, rTr, active, beta);
            k_update_p<<<(nPix + TPB - 1) / TPB, TPB, 0, stream>>>(pbuf, rbuf, beta, active);
        }
    }

    k_copy<<<(nPix + TPB - 1) / TPB, TPB, 0, stream>>>(out, xbuf, nPix);
}
